// Net_66949950210691
// MI455X (gfx1250) — matrix operand order repaired and checked by reading, not compiled, not run on hardware
//
#include <hip/hip_runtime.h>

// ---------------------------------------------------------------------------
// GCN (2x GCNConv + graph pooling) for MI455X / gfx1250.
// Dense transforms via v_wmma_f32_16x16x32_bf16 (wave32, 16x16 tiles).
// Edge aggregation via global_atomic_add_f32 (L2-resident scatter).
// ---------------------------------------------------------------------------

typedef __attribute__((ext_vector_type(16))) __bf16 v16bf;
typedef __attribute__((ext_vector_type(8)))  __bf16 v8bf;
typedef __attribute__((ext_vector_type(8)))  float  v8f;

#define SD   128   // embedding dim / conv1 in-features
#define F1   256   // conv1 out-features
#define NC   41    // num classes (conv2 out-features)
#define NCP  48    // NC padded to multiple of 16
#define NG   512   // num graphs

// ---------------------------------------------------------------- utilities
__global__ void k_fill_f32(float* __restrict__ p, float v, long n) {
    long i = (long)blockIdx.x * blockDim.x + threadIdx.x;
    long stride = (long)gridDim.x * blockDim.x;
    for (; i < n; i += stride) p[i] = v;
}

// deg[dst] += 1 for each real edge (deg pre-filled with 1.0 for self loop)
__global__ void k_degree(const int* __restrict__ edst, float* __restrict__ deg, int E) {
    long i = (long)blockIdx.x * blockDim.x + threadIdx.x;
    long stride = (long)gridDim.x * blockDim.x;
    for (; i < E; i += stride) atomicAdd(&deg[edst[i]], 1.0f);
}

__global__ void k_dinv(const float* __restrict__ deg, float* __restrict__ dinv, int N) {
    long i = (long)blockIdx.x * blockDim.x + threadIdx.x;
    long stride = (long)gridDim.x * blockDim.x;
    for (; i < N; i += stride) {
        float d = deg[i];
        dinv[i] = (d > 0.0f) ? __frsqrt_rn(d) : 0.0f;
    }
}

// norm[e] = dinv[src]*dinv[dst] for e < E; self loops (e >= E): dinv[n]^2
__global__ void k_norm(const int* __restrict__ esrc, const int* __restrict__ edst,
                       const float* __restrict__ dinv, float* __restrict__ norm,
                       int E, int N) {
    long i = (long)blockIdx.x * blockDim.x + threadIdx.x;
    long stride = (long)gridDim.x * blockDim.x;
    long total = (long)E + N;
    for (; i < total; i += stride) {
        int s, d;
        if (i < E) { s = esrc[i]; d = edst[i]; }
        else       { s = d = (int)(i - E); }
        norm[i] = dinv[s] * dinv[d];
    }
}

// xb[n][k] = bf16(emb[x_ids[n]][k])
__global__ void k_embed(const int* __restrict__ x_ids, const float* __restrict__ emb,
                        __bf16* __restrict__ xb, int N) {
    long i = (long)blockIdx.x * blockDim.x + threadIdx.x;
    long stride = (long)gridDim.x * blockDim.x;
    long total = (long)N * SD;
    for (; i < total; i += stride) {
        long n = i / SD, k = i % SD;
        xb[i] = (__bf16)emb[(long)x_ids[n] * SD + k];
    }
}

// W1t[o][k] = bf16(W1[k][o])   (W1: [SD, F1] row-major -> W1t: [F1, SD])
__global__ void k_convW1(const float* __restrict__ W1, __bf16* __restrict__ W1t) {
    long i = (long)blockIdx.x * blockDim.x + threadIdx.x;
    long stride = (long)gridDim.x * blockDim.x;
    const long total = (long)SD * F1;
    for (; i < total; i += stride) {
        long o = i / SD, k = i % SD;
        W1t[i] = (__bf16)W1[k * F1 + o];
    }
}

// W2t[o][k] = bf16(W2[k][o]) for o < NC, else 0   (W2: [F1, NC] -> W2t: [NCP, F1])
__global__ void k_convW2(const float* __restrict__ W2, __bf16* __restrict__ W2t) {
    long i = (long)blockIdx.x * blockDim.x + threadIdx.x;
    long stride = (long)gridDim.x * blockDim.x;
    const long total = (long)NCP * F1;
    for (; i < total; i += stride) {
        long o = i / F1, k = i % F1;
        W2t[i] = (o < NC) ? (__bf16)W2[k * NC + o] : (__bf16)0.0f;
    }
}

// -------------------------------------------------------------- WMMA GEMM
// C[M, Nout] = A[M, K] (bf16 row-major) x Bt[Nout, K]^T (bf16, pre-transposed)
// One wave32 per 16x16 output tile; K in steps of 32.
// A fragment (16-bit A 16x32 layout): lane m=lane&15, kh=lane>>4:
//   elems 0..7  -> K = k0 + kh*8 + j
//   elems 8..15 -> K = k0 + 16 + kh*8 + j
// B fragment (16-bit B 32x16): lane n=lane&15, kh=lane>>4:
//   elems 0..7  -> K = k0 + kh*8 + j, elems 8..15 -> K = k0 + 16 + kh*8 + j   (column n via Bt)
// C/D layout: VGPR r -> row (kh*8 + r), col = lane&15.
__global__ void k_gemm_bf16(const __bf16* __restrict__ A, const __bf16* __restrict__ Bt,
                            float* __restrict__ C, int M, int K, int Nout) {
    int wave = (int)((blockIdx.x * blockDim.x + threadIdx.x) >> 5);
    int lane = threadIdx.x & 31;
    int ntiles = Nout >> 4;
    int mt = wave / ntiles;
    int nt = wave - mt * ntiles;
    if (mt * 16 + 16 > M) return;

    int mr = (mt << 4) + (lane & 15);     // A row for this lane
    int nc = (nt << 4) + (lane & 15);     // B column for this lane
    int kh = lane >> 4;

    const __bf16* arow = A  + (long)mr * K;
    const __bf16* brow = Bt + (long)nc * K;

    v8f acc = {};
    for (int k0 = 0; k0 < K; k0 += 32) {
        // prefetch next K-slab (emits global_prefetch_b8)
        if (k0 + 32 < K) {
            __builtin_prefetch(arow + k0 + 32, 0, 0);
            __builtin_prefetch(brow + k0 + 32, 0, 0);
        }
        v8bf alo = *(const v8bf*)(arow + k0 + kh * 8);        // 16B aligned
        v8bf ahi = *(const v8bf*)(arow + k0 + 16 + kh * 8);   // 16B aligned
        v16bf a, b;
        v8bf blo = *(const v8bf*)(brow + k0 + kh * 8);        // 16B aligned
        v8bf bhi = *(const v8bf*)(brow + k0 + 16 + kh * 8);   // 16B aligned
#pragma unroll
        for (int j = 0; j < 8; ++j) { a[j] = alo[j]; a[8 + j] = ahi[j]; b[j] = blo[j]; b[8 + j] = bhi[j]; }
        acc = __builtin_amdgcn_wmma_f32_16x16x32_bf16(
            /*neg_a=*/false, a, /*neg_b=*/false, b,
            /*c_mod=*/(short)0, acc, /*reuse_a=*/false, /*reuse_b=*/false);
    }

    int row0 = (mt << 4) + (kh << 3);
    int col  = nc;
#pragma unroll
    for (int r = 0; r < 8; ++r)
        C[(long)(row0 + r) * Nout + col] = acc[r];
}

// ------------------------------------------------------- conv1 aggregation
// a1[dst] += h1[src] * norm  over all E+N entries, 4 floats per thread.
__global__ void k_scatter1(const int* __restrict__ esrc, const int* __restrict__ edst,
                           const float* __restrict__ norm, const float* __restrict__ h,
                           float* __restrict__ out, int E, int N) {
    const int chunks = F1 / 4;
    long total = ((long)E + N) * chunks;
    long i = (long)blockIdx.x * blockDim.x + threadIdx.x;
    long stride = (long)gridDim.x * blockDim.x;
    for (; i < total; i += stride) {
        int  c = (int)(i % chunks);
        long e = i / chunks;
        int s, d;
        if (e < E) { s = esrc[e]; d = edst[e]; }
        else       { s = d = (int)(e - E); }
        float w = norm[e];
        const float4 hv = *(const float4*)(h + (long)s * F1 + c * 4);
        float* o = out + (long)d * F1 + c * 4;
        atomicAdd(o + 0, hv.x * w);
        atomicAdd(o + 1, hv.y * w);
        atomicAdd(o + 2, hv.z * w);
        atomicAdd(o + 3, hv.w * w);
    }
}

// a1b = bf16(relu(a1 + b1))
__global__ void k_bias_relu(const float* __restrict__ a1, const float* __restrict__ b1,
                            __bf16* __restrict__ a1b, int N) {
    long total = (long)N * F1;
    long i = (long)blockIdx.x * blockDim.x + threadIdx.x;
    long stride = (long)gridDim.x * blockDim.x;
    for (; i < total; i += stride) {
        float v = a1[i] + b1[i % F1];
        a1b[i] = (__bf16)(v > 0.0f ? v : 0.0f);
    }
}

// ------------------------------------- conv2 aggregation fused with pooling
// out[batch[dst]][c] += h2[src][c] * norm[e]   (c < NC; h2 has NCP stride)
__global__ void k_scatter2_pool(const int* __restrict__ esrc, const int* __restrict__ edst,
                                const int* __restrict__ batch, const float* __restrict__ norm,
                                const float* __restrict__ h2, float* __restrict__ out,
                                int E, int N) {
    long total = ((long)E + N) * NC;
    long i = (long)blockIdx.x * blockDim.x + threadIdx.x;
    long stride = (long)gridDim.x * blockDim.x;
    for (; i < total; i += stride) {
        int  c = (int)(i % NC);
        long e = i / NC;
        int s, d;
        if (e < E) { s = esrc[e]; d = edst[e]; }
        else       { s = d = (int)(e - E); }
        float v = h2[(long)s * NCP + c] * norm[e];
        atomicAdd(&out[(long)batch[d] * NC + c], v);
    }
}

// out[batch[n]][c] += b2[c]   (per-node conv2 bias, pooled)
__global__ void k_bias2_pool(const int* __restrict__ batch, const float* __restrict__ b2,
                             float* __restrict__ out, int N) {
    long total = (long)N * NC;
    long i = (long)blockIdx.x * blockDim.x + threadIdx.x;
    long stride = (long)gridDim.x * blockDim.x;
    for (; i < total; i += stride) {
        int n = (int)(i / NC), c = (int)(i % NC);
        atomicAdd(&out[(long)batch[n] * NC + c], b2[c]);
    }
}

// ---------------------------------------------------------------------------
static inline int grid_for(long n, int block) {
    long b = (n + block - 1) / block;
    if (b < 1) b = 1;
    if (b > 1048576) b = 1048576;   // grid-stride loops cover the rest
    return (int)b;
}

extern "C" void kernel_launch(void* const* d_in, const int* in_sizes, int n_in,
                              void* d_out, int out_size, void* d_ws, size_t ws_size,
                              hipStream_t stream) {
    const int*   x_ids = (const int*)  d_in[0];
    const int*   eidx  = (const int*)  d_in[1];
    const int*   batch = (const int*)  d_in[2];
    const float* emb   = (const float*)d_in[3];
    const float* W1    = (const float*)d_in[4];
    const float* b1    = (const float*)d_in[5];
    const float* W2    = (const float*)d_in[6];
    const float* b2    = (const float*)d_in[7];
    float* out = (float*)d_out;

    const int N = in_sizes[0];
    const int E = in_sizes[1] / 2;
    const int* esrc = eidx;
    const int* edst = eidx + E;

    // --- workspace carve-out --------------------------------------------
    char* cur = (char*)d_ws;
    auto alloc = [&](size_t bytes) -> char* {
        char* p = cur;
        cur += (bytes + 255) & ~(size_t)255;
        return p;
    };
    float*  deg  = (float*) alloc((size_t)N * 4);
    float*  dinv = (float*) alloc((size_t)N * 4);
    float*  norm = (float*) alloc(((size_t)E + N) * 4);
    __bf16* xb   = (__bf16*)alloc((size_t)N * SD * 2);
    __bf16* W1t  = (__bf16*)alloc((size_t)F1 * SD * 2);
    __bf16* W2t  = (__bf16*)alloc((size_t)NCP * F1 * 2);
    float*  h1   = (float*) alloc((size_t)N * F1 * 4);
    float*  a1   = (float*) alloc((size_t)N * F1 * 4);
    __bf16* a1b  = (__bf16*)alloc((size_t)N * F1 * 2);
    float*  h2   = (float*) alloc((size_t)N * NCP * 4);
    (void)ws_size;

    const int BT = 256;

    // --- init ------------------------------------------------------------
    k_fill_f32<<<grid_for(N, BT), BT, 0, stream>>>(deg, 1.0f, (long)N);             // self loop
    k_fill_f32<<<grid_for((long)N * F1, BT), BT, 0, stream>>>(a1, 0.0f, (long)N * F1);
    k_fill_f32<<<grid_for((long)NG * NC, BT), BT, 0, stream>>>(out, 0.0f, (long)NG * NC);

    // --- degrees & norms -------------------------------------------------
    k_degree<<<grid_for(E, BT), BT, 0, stream>>>(edst, deg, E);
    k_dinv<<<grid_for(N, BT), BT, 0, stream>>>(deg, dinv, N);
    k_norm<<<grid_for((long)E + N, BT), BT, 0, stream>>>(esrc, edst, dinv, norm, E, N);

    // --- embedding gather + weight conversion ---------------------------
    k_embed<<<grid_for((long)N * SD, BT), BT, 0, stream>>>(x_ids, emb, xb, N);
    k_convW1<<<grid_for((long)SD * F1, BT), BT, 0, stream>>>(W1, W1t);
    k_convW2<<<grid_for((long)NCP * F1, BT), BT, 0, stream>>>(W2, W2t);

    // --- conv1: h1 = xb @ W1 (WMMA), scatter, bias+relu ------------------
    {
        long mtiles = ((long)N + 15) / 16;
        long waves  = mtiles * (F1 / 16);
        int blocks  = (int)((waves * 32 + BT - 1) / BT);
        k_gemm_bf16<<<blocks, BT, 0, stream>>>(xb, W1t, h1, N, SD, F1);
    }
    k_scatter1<<<grid_for(((long)E + N) * (F1 / 4), BT), BT, 0, stream>>>(
        esrc, edst, norm, h1, a1, E, N);
    k_bias_relu<<<grid_for((long)N * F1, BT), BT, 0, stream>>>(a1, b1, a1b, N);

    // --- conv2: h2 = a1b @ W2 (WMMA), fused scatter + pooling ------------
    {
        long mtiles = ((long)N + 15) / 16;
        long waves  = mtiles * (NCP / 16);
        int blocks  = (int)((waves * 32 + BT - 1) / BT);
        k_gemm_bf16<<<blocks, BT, 0, stream>>>(a1b, W2t, h2, N, F1, NCP);
    }
    k_scatter2_pool<<<grid_for(((long)E + N) * NC, BT), BT, 0, stream>>>(
        esrc, edst, batch, norm, h2, out, E, N);
    k_bias2_pool<<<grid_for((long)N * NC, BT), BT, 0, stream>>>(batch, b2, out, N);
}
